// MultiInputLSTM_57475252355462
// MI455X (gfx1250) — matrix operand order repaired and checked by reading, not compiled, not run on hardware
//
#include <hip/hip_runtime.h>
#include <hip/hip_fp16.h>

typedef __attribute__((ext_vector_type(16))) _Float16 v16h;
typedef __attribute__((ext_vector_type(8)))  _Float16 v8h;
typedef __attribute__((ext_vector_type(4)))  _Float16 v4h;
typedef __attribute__((ext_vector_type(8)))  float    v8f;

#define B_  128
#define T_  1024
#define I_  64
#define H_  128
#define K_  3
#define ROWS 16        // batch rows per workgroup (WMMA M granularity)
#define NTHREADS 256   // 8 waves of 32

// -------- LDS: only dynamic per-step data (~36KB). Weights live in VGPRs. --
struct Smem {
  _Float16 comb[4][ROWS][H_];      // [C_tilde, aux_p0..2]        16 KB
  _Float16 hsh[ROWS][H_];          // hidden state (f16)           4 KB
  _Float16 Ysh[2][ROWS][I_];       // Y_t ping-pong                4 KB
  _Float16 AXsh[2][K_][ROWS][I_];  // aux_t ping-pong             12 KB
};

// ---- WMMA fragment loaders (ISA 7.12.2 layouts, wave32) -------------------
// A (16-bit 16x32 MxK): lanes 0-15 row=lane, K in {kb+0..7, kb+16..23};
//                       lanes 16-31 row=lane-16, K in {kb+8..15, kb+24..31}
__device__ inline v16h load_a16(const _Float16* base, int stride, int lane, int kb) {
  const int row = lane & 15;
  const int kh  = (lane >> 4) << 3;
  const _Float16* p = base + row * stride + kb + kh;
  v8h lo = *(const v8h*)(p);          // 16B aligned -> ds_load_b128
  v8h hi = *(const v8h*)(p + 16);
  v16h a;
#pragma unroll
  for (int j = 0; j < 8; ++j) { a[j] = lo[j]; a[8 + j] = hi[j]; }
  return a;
}

// B (16-bit 32x16 KxN): column N = n0+(lane&15); K in {kb+8h+0..7, kb+16+8h+0..7}, h = lane>>4.
// One-time load straight from global fp32 weights, converted to f16.
__device__ inline v16h load_bfrag_global(const float* __restrict__ W, int ldw,
                                         int lane, int kb, int n0) {
  const float* p = W + (size_t)(kb + ((lane >> 4) << 3)) * ldw + n0 + (lane & 15);
  v16h b;
#pragma unroll
  for (int j = 0; j < 16; ++j) b[j] = (_Float16)p[(size_t)(j < 8 ? j : j + 8) * ldw];
  return b;
}

// reuse_a: identical consecutive WMMA sharing the A operand (gate groups)
// reuse_b: identical consecutive WMMA sharing the B operand (attention group)
__device__ inline v8f wmma_f16(v16h a, v16h b, v8f c) {
  return __builtin_amdgcn_wmma_f32_16x16x32_f16(false, a, false, b, (short)0, c, false, false);
}
__device__ inline v8f wmma_f16_ra(v16h a, v16h b, v8f c) {
  return __builtin_amdgcn_wmma_f32_16x16x32_f16(false, a, false, b, (short)0, c, true, false);
}
__device__ inline v8f wmma_f16_rb(v16h a, v16h b, v8f c) {
  return __builtin_amdgcn_wmma_f32_16x16x32_f16(false, a, false, b, (short)0, c, false, true);
}

// fast activations: v_exp_f32-based, saturate correctly, f16-consistent
__device__ inline float sigmoid_fast(float x) { return 1.0f / (1.0f + __expf(-x)); }
__device__ inline float tanh_fast(float x) {
  float e = __expf(2.0f * x);               // ->0 (x<<0), ->inf (x>>0)
  return 1.0f - 2.0f / (e + 1.0f);          // -> -1 / +1 at the limits
}

__device__ inline void st4h(_Float16* dst, float4 v) {
  v4h h = {(_Float16)v.x, (_Float16)v.y, (_Float16)v.z, (_Float16)v.w};
  *(v4h*)dst = h;   // 8B aligned
}

__global__ __launch_bounds__(NTHREADS, 1)
void milstm_scan_kernel(const float* __restrict__ Y, const float* __restrict__ aux,
                        const float* __restrict__ Wi, const float* __restrict__ Ui, const float* __restrict__ bi,
                        const float* __restrict__ Wf, const float* __restrict__ Uf, const float* __restrict__ bf,
                        const float* __restrict__ Wc, const float* __restrict__ Uc, const float* __restrict__ bc,
                        const float* __restrict__ Wo, const float* __restrict__ Uo, const float* __restrict__ bo,
                        const float* __restrict__ auxW, const float* __restrict__ auxb,
                        const float* __restrict__ Wa, const float* __restrict__ ba,
                        float* __restrict__ out) {
  __shared__ Smem s;
  const int tid   = threadIdx.x;
  const int lane  = tid & 31;
  const int wave  = tid >> 5;
  const int rbase = blockIdx.x * ROWS;    // this WG's 16 batch rows
  const int n0    = wave * 16;            // this wave's 16 output columns
  const int m_off = (lane >> 4) << 3;     // f32 frag: m = e + m_off
  const int ncol  = n0 + (lane & 15);     // f32 frag: n = ncol

  // ---- per-lane biases (registers) ---------------------------------------
  const float bg0 = bi[ncol], bg1 = bf[ncol], bg2 = bc[ncol], bg3 = bo[ncol];
  const float bas = ba[ncol];
  const float bax0 = auxb[0 * H_ + ncol], bax1 = auxb[1 * H_ + ncol],
              bax2 = auxb[2 * H_ + ncol];

  // ---- loop-invariant weight B-fragments -> registers (34 x v16h) --------
  v16h Ufr[4][4], Wgfr[4][2], AWfr[K_][2], Wafr[4];
  {
    const float* const Wsrc[4] = {Wi, Wf, Wc, Wo};
    const float* const Usrc[4] = {Ui, Uf, Uc, Uo};
#pragma unroll
    for (int g = 0; g < 4; ++g) {
#pragma unroll
      for (int kc = 0; kc < 4; ++kc)
        Ufr[g][kc] = load_bfrag_global(Usrc[g], H_, lane, kc * 32, n0);
#pragma unroll
      for (int kc = 0; kc < 2; ++kc)
        Wgfr[g][kc] = load_bfrag_global(Wsrc[g], H_, lane, kc * 32, n0);
    }
#pragma unroll
    for (int k = 0; k < K_; ++k)
#pragma unroll
      for (int kc = 0; kc < 2; ++kc)
        AWfr[k][kc] = load_bfrag_global(auxW + (size_t)k * I_ * H_, H_, lane, kc * 32, n0);
#pragma unroll
    for (int kc = 0; kc < 4; ++kc)
      Wafr[kc] = load_bfrag_global(Wa, H_, lane, kc * 32, n0);
  }

  // ---- zero h, stage t=0 inputs ------------------------------------------
  for (int idx = tid; idx < ROWS * H_; idx += NTHREADS)
    (&s.hsh[0][0])[idx] = (_Float16)0.0f;

  const int r  = tid >> 4;          // staging row (16 rows x 16 threads)
  const int i4 = (tid & 15) * 4;    // staging col (float4 granularity)
  const float* Yp  = Y + (size_t)(rbase + r) * T_ * I_ + i4;
  const float* Ap0 = aux + ((size_t)0 * B_ + rbase + r) * T_ * I_ + i4;
  const float* Ap1 = aux + ((size_t)1 * B_ + rbase + r) * T_ * I_ + i4;
  const float* Ap2 = aux + ((size_t)2 * B_ + rbase + r) * T_ * I_ + i4;

  {
    float4 y0 = *(const float4*)(Yp);
    float4 a0 = *(const float4*)(Ap0);
    float4 a1 = *(const float4*)(Ap1);
    float4 a2 = *(const float4*)(Ap2);
    st4h(&s.Ysh[0][r][i4], y0);
    st4h(&s.AXsh[0][0][r][i4], a0);
    st4h(&s.AXsh[0][1][r][i4], a1);
    st4h(&s.AXsh[0][2][r][i4], a2);
  }
  __syncthreads();

  // persistent per-wave state (f32 frag layout)
  v8f cst, hlast;
#pragma unroll
  for (int e = 0; e < 8; ++e) { cst[e] = 0.0f; hlast[e] = 0.0f; }

  for (int t = 0; t < T_; ++t) {
    const int cb = t & 1, nb = (t + 1) & 1;

    // ---- issue t+1 prefetch loads (consumed at loop bottom) --------------
    const int tn = (t + 1 < T_) ? (t + 1) : (T_ - 1);  // branchless clamp
    const size_t off = (size_t)tn * I_;
    float4 pY  = *(const float4*)(Yp + off);
    float4 pA0 = *(const float4*)(Ap0 + off);
    float4 pA1 = *(const float4*)(Ap1 + off);
    float4 pA2 = *(const float4*)(Ap2 + off);

    // ---- gate pre-activations: x-proj + recurrent (WMMA, C starts at 0) --
    v8f acc[4];
#pragma unroll
    for (int g = 0; g < 4; ++g)
#pragma unroll
      for (int e = 0; e < 8; ++e) acc[g][e] = 0.0f;

#pragma unroll
    for (int kc = 0; kc < 2; ++kc) {                 // Y_t @ W_*
      v16h afr = load_a16(&s.Ysh[cb][0][0], I_, lane, kc * 32);
      acc[0] = wmma_f16(afr, Wgfr[0][kc], acc[0]);
#pragma unroll
      for (int g = 1; g < 4; ++g) acc[g] = wmma_f16_ra(afr, Wgfr[g][kc], acc[g]);
    }
#pragma unroll
    for (int kc = 0; kc < 4; ++kc) {                 // h @ U_*
      v16h afr = load_a16(&s.hsh[0][0], H_, lane, kc * 32);
      acc[0] = wmma_f16(afr, Ufr[0][kc], acc[0]);
#pragma unroll
      for (int g = 1; g < 4; ++g) acc[g] = wmma_f16_ra(afr, Ufr[g][kc], acc[g]);
    }

    v8f aacc[K_];
#pragma unroll
    for (int k = 0; k < K_; ++k)
#pragma unroll
      for (int e = 0; e < 8; ++e) aacc[k][e] = 0.0f;
#pragma unroll
    for (int kc = 0; kc < 2; ++kc) {                 // aux_t @ aux_W
#pragma unroll
      for (int k = 0; k < K_; ++k) {
        v16h afr = load_a16(&s.AXsh[cb][k][0][0], I_, lane, kc * 32);
        aacc[k] = wmma_f16(afr, AWfr[k][kc], aacc[k]);
      }
    }

    // ---- activations (bias added here); publish comb for attention GEMM --
    v8f ig, fg, og, ctv, apv0, apv1, apv2;
#pragma unroll
    for (int e = 0; e < 8; ++e) {
      ig[e]   = sigmoid_fast(acc[0][e] + bg0);
      fg[e]   = sigmoid_fast(acc[1][e] + bg1);
      og[e]   = sigmoid_fast(acc[3][e] + bg3);
      ctv[e]  = tanh_fast(acc[2][e] + bg2);
      apv0[e] = tanh_fast(aacc[0][e] + bax0);
      apv1[e] = tanh_fast(aacc[1][e] + bax1);
      apv2[e] = tanh_fast(aacc[2][e] + bax2);
      const int m = e + m_off;
      s.comb[0][m][ncol] = (_Float16)ctv[e];
      s.comb[1][m][ncol] = (_Float16)apv0[e];
      s.comb[2][m][ncol] = (_Float16)apv1[e];
      s.comb[3][m][ncol] = (_Float16)apv2[e];
    }
    __syncthreads();

    // ---- attention logits: comb[j] @ W_a (WMMA, shared B -> reuse_b) -----
    v8f wacc[4];
#pragma unroll
    for (int j = 0; j < 4; ++j)
#pragma unroll
      for (int e = 0; e < 8; ++e) wacc[j][e] = 0.0f;
#pragma unroll
    for (int kc = 0; kc < 4; ++kc) {
      wacc[0] = wmma_f16(load_a16(&s.comb[0][0][0], H_, lane, kc * 32),
                         Wafr[kc], wacc[0]);
#pragma unroll
      for (int j = 1; j < 4; ++j)
        wacc[j] = wmma_f16_rb(load_a16(&s.comb[j][0][0], H_, lane, kc * 32),
                              Wafr[kc], wacc[j]);
    }

    // ---- softmax over 4 sources (values from registers), cell update -----
#pragma unroll
    for (int e = 0; e < 8; ++e) {
      const int m = e + m_off;
      float w0 = wacc[0][e] + bas, w1 = wacc[1][e] + bas,
            w2 = wacc[2][e] + bas, w3 = wacc[3][e] + bas;
      float mx = fmaxf(fmaxf(w0, w1), fmaxf(w2, w3));
      float e0 = __expf(w0 - mx), e1 = __expf(w1 - mx),
            e2 = __expf(w2 - mx), e3 = __expf(w3 - mx);
      float inv = 1.0f / (e0 + e1 + e2 + e3);
      float Lv  = (e0 * ctv[e] + e1 * apv0[e] + e2 * apv1[e] + e3 * apv2[e]) * inv;
      float cn  = fg[e] * cst[e] + ig[e] * Lv;
      cst[e]    = cn;
      float hn  = og[e] * tanh_fast(cn);
      hlast[e]  = hn;
      out[(size_t)B_ * H_ + ((size_t)(rbase + m) * T_ + t) * H_ + ncol] = hn;
      s.hsh[m][ncol] = (_Float16)hn;   // safe: all hsh reads were pre-mid-barrier
    }

    // ---- drain prefetch into the other ping-pong buffer ------------------
    st4h(&s.Ysh[nb][r][i4], pY);
    st4h(&s.AXsh[nb][0][r][i4], pA0);
    st4h(&s.AXsh[nb][1][r][i4], pA1);
    st4h(&s.AXsh[nb][2][r][i4], pA2);
    __syncthreads();
  }

  // ---- final h_t -> out[0 : B*H) -----------------------------------------
#pragma unroll
  for (int e = 0; e < 8; ++e)
    out[(size_t)(rbase + e + m_off) * H_ + ncol] = hlast[e];
}

extern "C" void kernel_launch(void* const* d_in, const int* in_sizes, int n_in,
                              void* d_out, int out_size, void* d_ws, size_t ws_size,
                              hipStream_t stream) {
  (void)in_sizes; (void)n_in; (void)out_size; (void)d_ws; (void)ws_size;
  const float* Y    = (const float*)d_in[0];
  const float* aux  = (const float*)d_in[1];
  const float* Wi   = (const float*)d_in[2];
  const float* Ui   = (const float*)d_in[3];
  const float* bi   = (const float*)d_in[4];
  const float* Wf   = (const float*)d_in[5];
  const float* Uf   = (const float*)d_in[6];
  const float* bf   = (const float*)d_in[7];
  const float* Wc   = (const float*)d_in[8];
  const float* Uc   = (const float*)d_in[9];
  const float* bc   = (const float*)d_in[10];
  const float* Wo   = (const float*)d_in[11];
  const float* Uo   = (const float*)d_in[12];
  const float* bo   = (const float*)d_in[13];
  const float* auxW = (const float*)d_in[14];
  const float* auxb = (const float*)d_in[15];
  const float* Wa   = (const float*)d_in[16];
  const float* ba   = (const float*)d_in[17];
  float* out = (float*)d_out;

  dim3 grid(B_ / ROWS);   // 8 independent batch blocks -> 8 WGPs
  dim3 block(NTHREADS);   // 8 waves
  milstm_scan_kernel<<<grid, block, 0, stream>>>(
      Y, aux, Wi, Ui, bi, Wf, Uf, bf, Wc, Uc, bc, Wo, Uo, bo,
      auxW, auxb, Wa, ba, out);
}
